// AngleUpdate_87978110091589
// MI455X (gfx1250) — matrix operand order repaired and checked by reading, not compiled, not run on hardware
//
#include <hip/hip_runtime.h>
#include <hip/hip_bf16.h>

// ---------------------------------------------------------------------------
// AngleUpdate on MI455X (gfx1250): bf16 WMMA 16x16x32, f32 accumulate.
// A-matrix = 16 angles x 256 features, B = pre-swizzled bf16 weights in LDS.
// ---------------------------------------------------------------------------

typedef __attribute__((ext_vector_type(16))) __bf16 v16bf;
typedef __attribute__((ext_vector_type(8)))  __bf16 v8bf;
typedef __attribute__((ext_vector_type(8)))  float  v8f;

#define N_ANGLES_C 1000000
#define KT1 8            // IN_D(256)/32
#define NT1 8            // HID_D(128)/16
#define KT2 4            // HID_D(128)/32
#define NT2 4            // OUT_D(64)/16
#define FRAG_HALFS 512   // 32 lanes * 16 bf16 per B fragment
#define W1_HALFS (KT1*NT1*FRAG_HALFS)   // 32768
#define W2_HALFS (KT2*NT2*FRAG_HALFS)   // 8192
#define WTOT_HALFS (2*(W1_HALFS + W2_HALFS))  // 81920 (160 KB)
#define HPITCH 136       // 128 + 8 pad (272B row stride -> no LDS bank collisions)
#define WAVES_PER_BLOCK 8

// --- prep: fp32 row-major W[K][N] -> bf16 B-fragments in wave32 WMMA layout.
// frag f = nt*KT + kt ; lane l<16: N=nt*16+l, K=kt*32+{0..7,16..23}
//                       lane l>=16: N=nt*16+l-16, K=kt*32+{8..15,24..31}
__global__ void swizzle_weights_kernel(const float* __restrict__ W,
                                       __bf16* __restrict__ dst,
                                       int KTt, int NTt) {
    int idx = blockIdx.x * blockDim.x + threadIdx.x;
    int total = KTt * NTt * FRAG_HALFS;
    if (idx >= total) return;
    int j  = idx & 15;
    int l  = (idx >> 4) & 31;
    int f  = idx >> 9;
    int kt = f % KTt;
    int nt = f / KTt;
    int n  = nt * 16 + (l & 15);
    int k  = kt * 32 + ((l < 16) ? 0 : 8) + (j & 7) + ((j >> 3) << 4);
    int ncols = NTt * 16;
    dst[idx] = (__bf16)W[(long long)k * ncols + n];
}

// One MLP (256->128 relu ->64) for a 16-angle tile. xf = 8 A-fragments.
// w1/w2 point at pre-swizzled bf16 fragments in LDS. z[4] = logits tiles.
__device__ __forceinline__ void run_mlp(const v16bf xf[KT1],
                                        const __bf16* __restrict__ w1,
                                        const float*  __restrict__ b1,
                                        const __bf16* __restrict__ w2,
                                        const float*  __restrict__ b2,
                                        __bf16* __restrict__ hbase,
                                        int lane, v8f z[NT2]) {
    const int m    = lane & 15;
    const int off8 = (lane < 16) ? 0 : 8;   // A-layout K sub-offset
    const int r0   = (lane < 16) ? 0 : 8;   // D-layout row base
    // ---- layer 1: h = relu(x @ W1 + b1), staged to LDS as bf16 ----
#pragma unroll
    for (int nt = 0; nt < NT1; ++nt) {
        float bv = b1[nt * 16 + m];
        v8f acc = {bv, bv, bv, bv, bv, bv, bv, bv};
#pragma unroll
        for (int kt = 0; kt < KT1; ++kt) {
            const v16bf* wp = (const v16bf*)(w1 + ((nt * KT1 + kt) * 32 + lane) * 16);
            acc = __builtin_amdgcn_wmma_f32_16x16x32_bf16(
                false, xf[kt], false, *wp, (short)0, acc, false, false);
        }
#pragma unroll
        for (int r = 0; r < 8; ++r) {
            float v = acc[r] > 0.0f ? acc[r] : 0.0f;
            hbase[(r0 + r) * HPITCH + nt * 16 + m] = (__bf16)v;
        }
    }
    // ---- reload h as layer-2 A fragments ----
    v16bf hf[KT2];
    const __bf16* hrow = hbase + m * HPITCH;
#pragma unroll
    for (int kt = 0; kt < KT2; ++kt) {
        int kb = kt * 32;
        v8bf c0 = *(const v8bf*)(hrow + kb + off8);
        v8bf c1 = *(const v8bf*)(hrow + kb + 16 + off8);
        hf[kt] = __builtin_shufflevector(c0, c1, 0, 1, 2, 3, 4, 5, 6, 7,
                                                 8, 9, 10, 11, 12, 13, 14, 15);
    }
    // ---- layer 2: z = h @ W2 + b2 ----
#pragma unroll
    for (int nt = 0; nt < NT2; ++nt) {
        float bv = b2[nt * 16 + m];
        v8f acc = {bv, bv, bv, bv, bv, bv, bv, bv};
#pragma unroll
        for (int kt = 0; kt < KT2; ++kt) {
            const v16bf* wp = (const v16bf*)(w2 + ((nt * KT2 + kt) * 32 + lane) * 16);
            acc = __builtin_amdgcn_wmma_f32_16x16x32_bf16(
                false, hf[kt], false, *wp, (short)0, acc, false, false);
        }
        z[nt] = acc;
    }
}

__launch_bounds__(256, 1)
__global__ void angle_update_kernel(const float* __restrict__ atom_feat,
                                    const float* __restrict__ bond_feat,
                                    const float* __restrict__ angle_feat,
                                    const int*   __restrict__ angle_index,
                                    const int*   __restrict__ edge_src,
                                    const int*   __restrict__ edge_dst,
                                    const float* __restrict__ gb1,
                                    const float* __restrict__ gb2,
                                    const float* __restrict__ ob1,
                                    const float* __restrict__ ob2,
                                    const __bf16* __restrict__ wsw,
                                    float* __restrict__ out,
                                    int n_tiles) {
    extern __shared__ __align__(16) unsigned char smem[];
    __bf16* wlds = (__bf16*)smem;
    // Cooperative copy of all pre-swizzled weights (160 KB bf16) into LDS.
    {
        const uint4* src = (const uint4*)wsw;
        uint4*       dst = (uint4*)wlds;
        const int nvec = WTOT_HALFS * (int)sizeof(__bf16) / 16;  // 10240
        for (int i = threadIdx.x; i < nvec; i += blockDim.x) dst[i] = src[i];
    }
    __syncthreads();

    const __bf16* gw1 = wlds;
    const __bf16* gw2 = wlds + W1_HALFS;
    const __bf16* ow1 = wlds + W1_HALFS + W2_HALFS;
    const __bf16* ow2 = wlds + 2 * W1_HALFS + W2_HALFS;

    const int lane = threadIdx.x & 31;
    const int wave = threadIdx.x >> 5;
    const int m    = lane & 15;
    const int off  = (lane < 16) ? 0 : 8;
    const int r0   = (lane < 16) ? 0 : 8;
    __bf16* hbase  = wlds + WTOT_HALFS + wave * (16 * HPITCH);

    for (int tile = blockIdx.x * WAVES_PER_BLOCK + wave; tile < n_tiles;
         tile += gridDim.x * WAVES_PER_BLOCK) {
        const long long a = (long long)tile * 16 + m;   // this lane's angle row
        const float* rows[4];
        rows[0] = bond_feat  + (long long)edge_src[a] * 64;
        rows[1] = bond_feat  + (long long)edge_dst[a] * 64;
        rows[2] = angle_feat + a * 64;
        rows[3] = atom_feat  + (long long)angle_index[a * 3 + 1] * 64;

        // Build the 8 A-fragments of x = [src_bond|dst_bond|angle|vertex].
        // A layout: lane m holds K {kb..kb+7, kb+16..kb+23}, lane m+16 the other halves.
        v16bf xf[KT1];
#pragma unroll
        for (int s = 0; s < 4; ++s) {
#pragma unroll
            for (int sub = 0; sub < 2; ++sub) {
                const int kb = sub * 32;
                const float4* p0 = (const float4*)(rows[s] + kb + off);
                const float4* p1 = (const float4*)(rows[s] + kb + 16 + off);
                float4 a0 = p0[0], a1 = p0[1], b0 = p1[0], b1 = p1[1];
                v16bf v;
                v[0]  = (__bf16)a0.x;  v[1]  = (__bf16)a0.y;
                v[2]  = (__bf16)a0.z;  v[3]  = (__bf16)a0.w;
                v[4]  = (__bf16)a1.x;  v[5]  = (__bf16)a1.y;
                v[6]  = (__bf16)a1.z;  v[7]  = (__bf16)a1.w;
                v[8]  = (__bf16)b0.x;  v[9]  = (__bf16)b0.y;
                v[10] = (__bf16)b0.z;  v[11] = (__bf16)b0.w;
                v[12] = (__bf16)b1.x;  v[13] = (__bf16)b1.y;
                v[14] = (__bf16)b1.z;  v[15] = (__bf16)b1.w;
                xf[s * 2 + sub] = v;
            }
        }

        v8f zg[NT2], zo[NT2];
        run_mlp(xf, gw1, gb1, gw2, gb2, hbase, lane, zg);  // gate logits
        run_mlp(xf, ow1, ob1, ow2, ob2, hbase, lane, zo);  // out  logits

        // result = angle_feat + silu(zo) * sigmoid(zg); coalesced row stores.
#pragma unroll
        for (int nt = 0; nt < NT2; ++nt) {
            const int col = nt * 16 + m;
#pragma unroll
            for (int r = 0; r < 8; ++r) {
                const long long arow = (long long)tile * 16 + r0 + r;
                float g  = 1.0f / (1.0f + __expf(-zg[nt][r]));
                float zz = zo[nt][r];
                float o  = zz / (1.0f + __expf(-zz));
                out[arow * 64 + col] = angle_feat[arow * 64 + col] + o * g;
            }
        }
    }
}

extern "C" void kernel_launch(void* const* d_in, const int* in_sizes, int n_in,
                              void* d_out, int out_size, void* d_ws, size_t ws_size,
                              hipStream_t stream) {
    const float* atom_feat  = (const float*)d_in[0];
    const float* bond_feat  = (const float*)d_in[1];
    const float* angle_feat = (const float*)d_in[2];
    const int*   angle_idx  = (const int*)d_in[3];
    const int*   edge_src   = (const int*)d_in[4];
    const int*   edge_dst   = (const int*)d_in[5];
    const float* gW1 = (const float*)d_in[6];
    const float* gb1 = (const float*)d_in[7];
    const float* gW2 = (const float*)d_in[8];
    const float* gb2 = (const float*)d_in[9];
    const float* oW1 = (const float*)d_in[10];
    const float* ob1 = (const float*)d_in[11];
    const float* oW2 = (const float*)d_in[12];
    const float* ob2 = (const float*)d_in[13];
    float*  out = (float*)d_out;
    __bf16* wsw = (__bf16*)d_ws;   // 160 KB of pre-swizzled bf16 weights

    // Pre-swizzle weights into WMMA B-fragment order every call (deterministic).
    swizzle_weights_kernel<<<(W1_HALFS + 255) / 256, 256, 0, stream>>>(
        gW1, wsw, KT1, NT1);
    swizzle_weights_kernel<<<(W2_HALFS + 255) / 256, 256, 0, stream>>>(
        gW2, wsw + W1_HALFS, KT2, NT2);
    swizzle_weights_kernel<<<(W1_HALFS + 255) / 256, 256, 0, stream>>>(
        oW1, wsw + W1_HALFS + W2_HALFS, KT1, NT1);
    swizzle_weights_kernel<<<(W2_HALFS + 255) / 256, 256, 0, stream>>>(
        oW2, wsw + 2 * W1_HALFS + W2_HALFS, KT2, NT2);

    const int n_tiles = N_ANGLES_C / 16;  // 62500
    const size_t smem =
        (size_t)(WTOT_HALFS + WAVES_PER_BLOCK * 16 * HPITCH) * sizeof(__bf16);
    (void)hipFuncSetAttribute((const void*)angle_update_kernel,
                              hipFuncAttributeMaxDynamicSharedMemorySize,
                              (int)smem);
    const int grid = 768;  // grid-stride over 62500 tiles, 8 waves/block
    angle_update_kernel<<<grid, 256, smem, stream>>>(
        atom_feat, bond_feat, angle_feat, angle_idx, edge_src, edge_dst,
        gb1, gb2, ob1, ob2, wsw, out, n_tiles);
}
